// TrajectoryPredictor_12360915877917
// MI455X (gfx1250) — compile-verified, matrix operand order checked by reading, not run on hardware
//
#include <hip/hip_runtime.h>
#include <hip/hip_bf16.h>
#include <math.h>

typedef __hip_bfloat16 bf16;
typedef __attribute__((ext_vector_type(16))) __bf16 v16bf;
typedef __attribute__((ext_vector_type(8)))  float  v8f;

#define DEV __device__ __forceinline__
#define Z8 {0.f,0.f,0.f,0.f,0.f,0.f,0.f,0.f}

DEV float eluf(float x){ return x > 0.f ? x : __expf(x) - 1.f; }
DEV float sigf(float x){ return 1.f / (1.f + __expf(-x)); }
DEV float clipf(float x){ return fminf(fmaxf(x, -5.f), 2.f); }

// ---- WMMA fragment load: 16x32 bf16 tile, A-operand layout (also used for B
// via pre-transposed W^T). Lane l%16 = row, two contiguous 16B chunks:
// K offsets [8*(l/16), +8) and [16+8*(l/16), +8).
DEV v16bf frag_ld(const bf16* p, int stride){
  const int lane = threadIdx.x & 31;
  const bf16* q = p + (size_t)(lane & 15) * stride + ((lane >> 4) << 3);
  union { uint4 u[2]; v16bf v; } cv;
  cv.u[0] = *(const uint4*)(q);
  cv.u[1] = *(const uint4*)(q + 16);
  return cv.v;
}

DEV v8f wmma_bf16(v16bf a, v16bf b, v8f c){
  return __builtin_amdgcn_wmma_f32_16x16x32_bf16(false, a, false, b, (short)0, c, false, false);
}

// Fully-unrolled, 2-deep software-pipelined K loop: next frags are fetched
// before the current wmma consumes its operands, so the inserted
// s_wait_loadcnt/dscnt covers only the older pair (non-zero wait -> overlap).
template<int KITERS>
DEV v8f gemm16(const bf16* A, int lda, const bf16* W, int ldw, v8f acc){
  v16bf a = frag_ld(A, lda);
  v16bf b = frag_ld(W, ldw);
  #pragma unroll
  for (int i = 1; i < KITERS; ++i){
    v16bf a2 = frag_ld(A + i * 32, lda);
    v16bf b2 = frag_ld(W + i * 32, ldw);
    acc = wmma_bf16(a, b, acc);
    a = a2; b = b2;
  }
  return wmma_bf16(a, b, acc);
}

// C/D layout: VGPR r -> row = r + 8*(lane/16), col = lane%16
DEV void st_bf16(v8f c, bf16* D, int ldd, const float* bias, int nbias, int n0, int act){
  const int lane = threadIdx.x & 31;
  const int col = lane & 15, rb = (lane >> 4) << 3;
  float b = (bias && (n0 + col) < nbias) ? bias[n0 + col] : 0.f;
  #pragma unroll
  for (int r = 0; r < 8; ++r){
    float v = c[r] + b;
    if (act == 1) v = eluf(v);
    D[(size_t)(rb + r) * ldd + col] = __float2bfloat16(v);
  }
}

DEV void st_f32(v8f c, float* D, int ldd, const float* b1, int nb1,
                const float* b2, int n0, int act){
  const int lane = threadIdx.x & 31;
  const int col = lane & 15, rb = (lane >> 4) << 3;
  float b = 0.f;
  if (b1 && (n0 + col) < nb1) b += b1[n0 + col];
  if (b2) b += b2[n0 + col];
  #pragma unroll
  for (int r = 0; r < 8; ++r){
    float v = c[r] + b;
    if (act == 2) v = sigf(v);
    D[(rb + r) * ldd + col] = v;
  }
}

// Y[16 x N] = act(A[16 x KP] @ W^T(NxKP) + bias), N-tiles split across 4 waves.
template<int KP>
DEV void layer_bf16(const bf16* A, int lda, const bf16* W, const float* bias,
                    int N, bf16* D, int ldd, int act, int wave){
  for (int t = wave; t < (N >> 4); t += 4){
    v8f acc = Z8;
    acc = gemm16<KP/32>(A, lda, W + (size_t)(t * 16) * KP, KP, acc);
    st_bf16(acc, D + t * 16, ldd, bias, N, t * 16, act);
  }
}

template<int KP>
DEV void layer_f32(const bf16* A, int lda, const bf16* W, const float* bias,
                   int N, float* D, int ldd, int wave){
  for (int t = wave; t < (N >> 4); t += 4){
    v8f acc = Z8;
    acc = gemm16<KP/32>(A, lda, W + (size_t)(t * 16) * KP, KP, acc);
    st_f32(acc, D + t * 16, ldd, bias, N, nullptr, t * 16, 0);
  }
}

// One GRU cell on a 16-row tile. Hidden 256, processed in two 128-wide chunks.
// r,z gates fuse x@Wih + h@Whh in one accumulator; n gate kept separate.
template<int KX, int LDW>
DEV void gru_layer(const bf16* X, int ldx, const bf16* Wih,
                   const bf16* H, const bf16* Whh,
                   const float* bih, const float* bhh, bf16* Hnew,
                   float* sFa, float* sFb, float* sFc, float* sFd,
                   int wave, int tid){
  for (int c0 = 0; c0 < 256; c0 += 128){
    for (int tl = wave; tl < 8; tl += 4){                    // r gate -> sFc
      int ng = c0 + tl * 16;
      v8f acc = Z8;
      acc = gemm16<KX>(X, ldx, Wih + (size_t)ng * LDW, LDW, acc);
      acc = gemm16<8>(H, 256, Whh + (size_t)ng * 256, 256, acc);
      st_f32(acc, sFc + tl * 16, 128, bih, 1 << 30, bhh, ng, 2);
    }
    for (int tl = wave; tl < 8; tl += 4){                    // z gate -> sFd
      int ng = 256 + c0 + tl * 16;
      v8f acc = Z8;
      acc = gemm16<KX>(X, ldx, Wih + (size_t)ng * LDW, LDW, acc);
      acc = gemm16<8>(H, 256, Whh + (size_t)ng * 256, 256, acc);
      st_f32(acc, sFd + tl * 16, 128, bih, 1 << 30, bhh, ng, 2);
    }
    for (int tl = wave; tl < 8; tl += 4){                    // i_n -> sFa, hn -> sFb
      int ng = 512 + c0 + tl * 16;
      v8f a1 = Z8;
      a1 = gemm16<KX>(X, ldx, Wih + (size_t)ng * LDW, LDW, a1);
      st_f32(a1, sFa + tl * 16, 128, bih, 1 << 30, nullptr, ng, 0);
      v8f a2 = Z8;
      a2 = gemm16<8>(H, 256, Whh + (size_t)ng * 256, 256, a2);
      st_f32(a2, sFb + tl * 16, 128, bhh, 1 << 30, nullptr, ng, 0);
    }
    __syncthreads();
    for (int i = tid; i < 16 * 128; i += 128){
      int r = i >> 7, c = i & 127;
      float nn = tanhf(sFa[i] + sFc[i] * sFb[i]);
      float zg = sFd[i];
      float ho = __bfloat162float(H[r * 256 + c0 + c]);
      Hnew[r * 256 + c0 + c] = __float2bfloat16((1.f - zg) * nn + zg * ho);
    }
    __syncthreads();
  }
}

struct KParams {
  const float *past, *future, *ctx, *world, *eps;
  const float *b_se1,*b_se2,*b_q,*b_k,*b_v,*b_o,*b_we1,*b_we2;
  const float *bih0,*bhh0,*bih1,*bhh1;
  const float *b_post1,*b_post2,*b_prior1,*b_prior2,*b_dec1,*b_dec2,*b_mh,*b_lh;
  const float *ln_g,*ln_b;
  const bf16 *Wse1,*Wse2,*Wq,*Wk,*Wv,*Wo,*Wwe1,*Wwe2;
  const bf16 *Wih0,*Whh0,*Wih1,*Whh1;
  const bf16 *Wpost1,*Wpost2,*Wprior1,*Wprior2,*Wdec1,*Wdec2,*Wmh,*Wlh;
  float* out;
  int B;
};

static constexpr int SMEM_BYTES = 131648;

__global__ __launch_bounds__(128) void traj_kernel(KParams P){
  const int tid = threadIdx.x, wave = tid >> 5;
  const int b0 = blockIdx.x * 16;
  extern __shared__ char smem_raw[];
  bf16* sIn  = (bf16*)smem_raw;          // 16x384 (x / post-in / dec-in)
  bf16* sEmb = sIn  + 16*384;            // 16x128
  bf16* sHid = sEmb + 16*128;            // 16x256 (av / layer hiddens)
  bf16* sT   = sHid + 16*256;            // 16x128
  bf16* sPS  = sT   + 16*128;            // 16x32 padded state
  bf16* sPW  = sPS  + 16*32;             // 16x32 padded world
  bf16* sH0  = sPW  + 16*32;             // 16x256
  bf16* sH1  = sH0  + 16*256;            // 16x256
  bf16* sHn  = sH1  + 16*256;            // 16x256 staging
  bf16* sZ   = sHn  + 16*256;            // 16x64
  bf16* sCtx = sZ   + 16*64;             // 128x32 padded neighbor tokens
  bf16* sKm  = sCtx + 128*32;            // 128x64
  bf16* sVm  = sKm  + 128*64;            // 128x64
  float* sCn = (float*)(sVm + 128*64);   // 128 neighbor norms^2
  float* sKl = sCn + 128;                // 16
  float* sFa = sKl + 16;                 // 16x128 f32 scratch x4
  float* sFb = sFa + 16*128;
  float* sFc = sFb + 16*128;
  float* sFd = sFc + 16*128;

  for (int i = tid; i < 16*256; i += 128){ sH0[i] = __float2bfloat16(0.f); sH1[i] = __float2bfloat16(0.f); }
  for (int i = tid; i < 16*64;  i += 128)  sZ[i]  = __float2bfloat16(0.f);
  __syncthreads();

  for (int t = 0; t < 40; ++t){
    // ---- phase 1: stage inputs (padded bf16) ----
    for (int i = tid; i < 16*32; i += 128){
      int r = i >> 5, c = i & 31, b = b0 + r;
      float sv = 0.f, wv = 0.f;
      if (c < 10) sv = (t < 20) ? P.past[((size_t)b*20 + t)*10 + c]
                                : P.future[((size_t)b*20 + (t-20))*10 + c];
      if (c < 12) wv = P.world[((size_t)b*40 + t)*12 + c];
      sPS[i] = __float2bfloat16(sv);
      sPW[i] = __float2bfloat16(wv);
    }
    for (int i = tid; i < 128*32; i += 128){
      int tok = i >> 5, c = i & 31;
      int r = tok >> 3, n = tok & 7, b = b0 + r;
      float v = (c < 10) ? P.ctx[(((size_t)b*40 + t)*8 + n)*10 + c] : 0.f;
      sCtx[i] = __float2bfloat16(v);
    }
    {
      int r = tid >> 3, n = tid & 7, b = b0 + r;
      const float* cp = &P.ctx[(((size_t)b*40 + t)*8 + n)*10];
      float s2 = 0.f;
      for (int d = 0; d < 10; ++d) s2 += cp[d]*cp[d];
      sCn[tid] = s2;
    }
    __syncthreads();

    // ---- state encoder ----
    layer_bf16<32>(sPS, 32, P.Wse1, P.b_se1, 128, sT, 128, 1, wave);
    __syncthreads();
    layer_bf16<128>(sT, 128, P.Wse2, P.b_se2, 128, sEmb, 128, 1, wave);
    __syncthreads();

    // ---- attention projections: Q (f32), K/V (bf16) ----
    layer_f32<128>(sEmb, 128, P.Wq, P.b_q, 64, sFa, 64, wave);
    for (int tt = wave; tt < 32; tt += 4){
      int m = tt >> 2, nt = tt & 3;
      v8f a1 = Z8;
      a1 = gemm16<1>(sCtx + m*16*32, 32, P.Wk + (size_t)(nt*16)*32, 32, a1);
      st_bf16(a1, sKm + m*16*64 + nt*16, 64, P.b_k, 64, nt*16, 0);
      v8f a2 = Z8;
      a2 = gemm16<1>(sCtx + m*16*32, 32, P.Wv + (size_t)(nt*16)*32, 32, a2);
      st_bf16(a2, sVm + m*16*64 + nt*16, 64, P.b_v, 64, nt*16, 0);
    }
    __syncthreads();

    // ---- softmax attention (16 rows x 2 heads on 32 lanes) ----
    if (tid < 32){
      int r = tid & 15, h = tid >> 4;
      const float scale = 0.17677669529663687f;  // 1/sqrt(32)
      float sc[8], mx = -1e30f;
      for (int n = 0; n < 8; ++n){
        float s = 0.f;
        const bf16* kp = sKm + (r*8 + n)*64 + h*32;
        const float* qp = sFa + r*64 + h*32;
        for (int d = 0; d < 32; ++d) s += qp[d] * __bfloat162float(kp[d]);
        s *= scale;
        if (sCn[r*8 + n] < 1e-12f) s = -1e9f;
        sc[n] = s; mx = fmaxf(mx, s);
      }
      float den = 0.f;
      for (int n = 0; n < 8; ++n){ sc[n] = __expf(sc[n]-mx); den += sc[n]; }
      float inv = 1.f / den;
      for (int d = 0; d < 32; ++d){
        float av = 0.f;
        for (int n = 0; n < 8; ++n)
          av += sc[n] * __bfloat162float(sVm[(r*8+n)*64 + h*32 + d]);
        sHid[r*64 + h*32 + d] = __float2bfloat16(av * inv);
      }
    }
    __syncthreads();
    layer_f32<64>(sHid, 64, P.Wo, P.b_o, 64, sFb, 64, wave);
    __syncthreads();

    // ---- layernorm -> x[128:192]; copy emb -> x[0:128], z -> x[192:256] ----
    if (tid < 16){
      float mu = 0.f;
      for (int j = 0; j < 64; ++j) mu += sFb[tid*64+j];
      mu *= (1.f/64.f);
      float var = 0.f;
      for (int j = 0; j < 64; ++j){ float d = sFb[tid*64+j]-mu; var += d*d; }
      var *= (1.f/64.f);
      float rs = rsqrtf(var + 1e-5f);
      for (int j = 0; j < 64; ++j){
        float y = (sFb[tid*64+j]-mu)*rs*P.ln_g[j] + P.ln_b[j];
        sIn[tid*384 + 128 + j] = __float2bfloat16(y);
      }
    } else {
      for (int i = tid - 16; i < 16*128; i += 112){
        int r = i >> 7, c = i & 127;
        sIn[r*384 + c] = sEmb[r*128 + c];
      }
      for (int i = tid - 16; i < 16*64; i += 112){
        int r = i >> 6, c = i & 63;
        sIn[r*384 + 192 + c] = sZ[r*64 + c];
      }
    }
    __syncthreads();

    // ---- world embed -> x[256:320] ----
    layer_bf16<32>(sPW, 32, P.Wwe1, P.b_we1, 64, sT, 64, 1, wave);
    __syncthreads();
    layer_bf16<64>(sT, 64, P.Wwe2, P.b_we2, 64, sIn + 256, 384, 1, wave);
    __syncthreads();

    // ---- GRU stack ----
    gru_layer<10, 320>(sIn, 384, P.Wih0, sH0, P.Whh0, P.bih0, P.bhh0, sHn,
                       sFa, sFb, sFc, sFd, wave, tid);
    for (int i = tid; i < 16*256; i += 128) sH0[i] = sHn[i];
    __syncthreads();
    gru_layer<8, 256>(sH0, 256, P.Wih1, sH1, P.Whh1, P.bih1, P.bhh1, sHn,
                      sFa, sFb, sFc, sFd, wave, tid);
    for (int i = tid; i < 16*256; i += 128) sH1[i] = sHn[i];
    __syncthreads();

    // ---- posterior: in = [ht | emb] ----
    for (int i = tid; i < 16*256; i += 128){ int r = i >> 8, c = i & 255; sIn[r*384 + c] = sH1[i]; }
    for (int i = tid; i < 16*128; i += 128){ int r = i >> 7, c = i & 127; sIn[r*384 + 256 + c] = sEmb[i]; }
    __syncthreads();
    layer_bf16<384>(sIn, 384, P.Wpost1, P.b_post1, 256, sHid, 256, 1, wave);
    __syncthreads();
    layer_f32<256>(sHid, 256, P.Wpost2, P.b_post2, 128, sFa, 128, wave);
    __syncthreads();
    // ---- prior ----
    layer_bf16<256>(sH1, 256, P.Wprior1, P.b_prior1, 256, sHid, 256, 1, wave);
    __syncthreads();
    layer_f32<256>(sHid, 256, P.Wprior2, P.b_prior2, 128, sFb, 128, wave);
    __syncthreads();

    // ---- KL + reparameterize ----
    if (tid < 16){
      float kl = 0.f;
      const float* ep = P.eps + ((size_t)t * P.B + b0 + tid) * 64;
      for (int j = 0; j < 64; ++j){
        float pm  = sFa[tid*128 + j];
        float pls = clipf(sFa[tid*128 + 64 + j]);
        float prm = sFb[tid*128 + j];
        float prls= clipf(sFb[tid*128 + 64 + j]);
        float dm = pm - prm;
        kl += prls - pls + (__expf(2.f*pls) + dm*dm) / (2.f*__expf(2.f*prls)) - 0.5f;
        sZ[tid*64 + j] = __float2bfloat16(pm + __expf(pls) * ep[j]);
      }
      sKl[tid] = kl;
    }
    __syncthreads();

    // ---- decoder: in = [ht | zn] (ht already in sIn[0:256]) ----
    for (int i = tid; i < 16*64; i += 128){ int r = i >> 6, c = i & 63; sIn[r*384 + 256 + c] = sZ[i]; }
    __syncthreads();
    layer_bf16<320>(sIn, 384, P.Wdec1, P.b_dec1, 256, sHid, 256, 1, wave);
    __syncthreads();
    layer_bf16<256>(sHid, 256, P.Wdec2, P.b_dec2, 128, sT, 128, 1, wave);
    __syncthreads();

    // ---- output heads (N padded 10 -> 16) ----
    if (wave == 0){
      v8f acc = Z8;
      acc = gemm16<4>(sT, 128, P.Wmh, 128, acc);
      st_f32(acc, sFc, 16, P.b_mh, 10, nullptr, 0, 0);
    } else if (wave == 1){
      v8f acc = Z8;
      acc = gemm16<4>(sT, 128, P.Wlh, 128, acc);
      st_f32(acc, sFd, 16, P.b_lh, 10, nullptr, 0, 0);
    }
    __syncthreads();
    if (tid < 16){
      float* op = P.out + ((size_t)(b0 + tid)*40 + t)*21;
      for (int j = 0; j < 10; ++j) op[j]      = sFc[tid*16 + j];
      for (int j = 0; j < 10; ++j) op[10 + j] = clipf(sFd[tid*16 + j]);
      op[20] = sKl[tid];
    }
    __syncthreads();
  }
}

// ---- weight prep: f32 (din,dout) -> bf16 W^T (doutp, dinp), zero-padded ----
__global__ void prep_t(const float* src, bf16* dst, int din, int dout, int dinp, int doutp){
  int i = blockIdx.x * 256 + threadIdx.x;
  if (i >= doutp * dinp) return;
  int n = i / dinp, k = i % dinp;
  float v = (n < dout && k < din) ? src[(size_t)k * dout + n] : 0.f;
  dst[i] = __float2bfloat16(v);
}
// ---- GRU weights already (out,in) row-major: convert + pad cols ----
__global__ void prep_c(const float* src, bf16* dst, int rows, int din, int dinp){
  int i = blockIdx.x * 256 + threadIdx.x;
  if (i >= rows * dinp) return;
  int r = i / dinp, c = i % dinp;
  dst[i] = __float2bfloat16(c < din ? src[(size_t)r * din + c] : 0.f);
}

extern "C" void kernel_launch(void* const* d_in, const int* in_sizes, int n_in,
                              void* d_out, int out_size, void* d_ws, size_t ws_size,
                              hipStream_t stream){
  (void)out_size; (void)ws_size;
  auto F = [&](int i){ return (const float*)d_in[i]; };
  const int B = in_sizes[0] / 200;   // past_states: B*20*10
  const int epsIdx = n_in - 1;

  bf16* w = (bf16*)d_ws;
  size_t off = 0;
  auto alloc = [&](size_t n){ bf16* p = w + off; off += n; return p; };
  bf16 *Wse1 = alloc(128*32),  *Wse2 = alloc(128*128);
  bf16 *Wq   = alloc(64*128),  *Wk   = alloc(64*32), *Wv = alloc(64*32);
  bf16 *Wo   = alloc(64*64),   *Wwe1 = alloc(64*32), *Wwe2 = alloc(64*64);
  bf16 *Wih0 = alloc(768*320), *Whh0 = alloc(768*256);
  bf16 *Wih1 = alloc(768*256), *Whh1 = alloc(768*256);
  bf16 *Wpost1 = alloc(256*384), *Wpost2 = alloc(128*256);
  bf16 *Wprior1 = alloc(256*256), *Wprior2 = alloc(128*256);
  bf16 *Wdec1 = alloc(256*320), *Wdec2 = alloc(128*256);
  bf16 *Wmh = alloc(16*128), *Wlh = alloc(16*128);

  auto T = [&](const float* s, bf16* d, int din, int dout, int dinp, int doutp){
    int n = doutp * dinp;
    prep_t<<<(n + 255)/256, 256, 0, stream>>>(s, d, din, dout, dinp, doutp);
  };
  auto C = [&](const float* s, bf16* d, int rows, int din, int dinp){
    int n = rows * dinp;
    prep_c<<<(n + 255)/256, 256, 0, stream>>>(s, d, rows, din, dinp);
  };
  // params pytree (sorted keys): dec1,dec2,gru0,gru1,k,lh,ln_b,ln_g,mh,o,
  // post1,post2,prior1,prior2,q,se1,se2,v,we1,we2 -> d_in[4..45]
  T(F(36), Wse1, 10, 128, 32, 128);    T(F(38), Wse2, 128, 128, 128, 128);
  T(F(34), Wq, 128, 64, 128, 64);      T(F(16), Wk, 10, 64, 32, 64);
  T(F(40), Wv, 10, 64, 32, 64);        T(F(24), Wo, 64, 64, 64, 64);
  T(F(42), Wwe1, 12, 64, 32, 64);      T(F(44), Wwe2, 64, 64, 64, 64);
  C(F(11), Wih0, 768, 320, 320);       C(F(10), Whh0, 768, 256, 256);
  C(F(15), Wih1, 768, 256, 256);       C(F(14), Whh1, 768, 256, 256);
  T(F(26), Wpost1, 384, 256, 384, 256);T(F(28), Wpost2, 256, 128, 256, 128);
  T(F(30), Wprior1, 256, 256, 256, 256);T(F(32), Wprior2, 256, 128, 256, 128);
  T(F(4),  Wdec1, 320, 256, 320, 256); T(F(6),  Wdec2, 256, 128, 256, 128);
  T(F(22), Wmh, 128, 10, 128, 16);     T(F(18), Wlh, 128, 10, 128, 16);

  KParams P;
  P.past = F(0); P.future = F(1); P.ctx = F(2); P.world = F(3); P.eps = F(epsIdx);
  P.b_se1 = F(37); P.b_se2 = F(39); P.b_q = F(35); P.b_k = F(17); P.b_v = F(41);
  P.b_o = F(25); P.b_we1 = F(43); P.b_we2 = F(45);
  P.bih0 = F(9); P.bhh0 = F(8); P.bih1 = F(13); P.bhh1 = F(12);
  P.b_post1 = F(27); P.b_post2 = F(29); P.b_prior1 = F(31); P.b_prior2 = F(33);
  P.b_dec1 = F(5); P.b_dec2 = F(7); P.b_mh = F(23); P.b_lh = F(19);
  P.ln_g = F(21); P.ln_b = F(20);
  P.Wse1 = Wse1; P.Wse2 = Wse2; P.Wq = Wq; P.Wk = Wk; P.Wv = Wv; P.Wo = Wo;
  P.Wwe1 = Wwe1; P.Wwe2 = Wwe2;
  P.Wih0 = Wih0; P.Whh0 = Whh0; P.Wih1 = Wih1; P.Whh1 = Whh1;
  P.Wpost1 = Wpost1; P.Wpost2 = Wpost2; P.Wprior1 = Wprior1; P.Wprior2 = Wprior2;
  P.Wdec1 = Wdec1; P.Wdec2 = Wdec2; P.Wmh = Wmh; P.Wlh = Wlh;
  P.out = (float*)d_out; P.B = B;

  hipFuncSetAttribute((const void*)traj_kernel,
                      hipFuncAttributeMaxDynamicSharedMemorySize, SMEM_BYTES);
  traj_kernel<<<dim3(B / 16), dim3(128), SMEM_BYTES, stream>>>(P);
}
